// LexicalizedGrammarDecoder_52647709114848
// MI455X (gfx1250) — matrix operand order repaired and checked by reading, not compiled, not run on hardware
//
#include <hip/hip_runtime.h>
#include <hip/hip_bf16.h>
#include <cstdint>

typedef unsigned short u16;
typedef __attribute__((ext_vector_type(16))) __bf16    v16bf;
typedef __attribute__((ext_vector_type(8)))  float     v8f;
typedef __attribute__((ext_vector_type(4)))  unsigned  v4u;
typedef __attribute__((ext_vector_type(8)))  unsigned  v8u;

// ---------------- problem constants ----------------
constexpr int B      = 128;
constexpr int S_SRC  = 128;
constexpr int S_TRG  = 64;
constexpr int LEX_D  = 300;
constexpr int LDP    = 320;   // LEX_D padded to x32
constexpr int NT_D   = 128;
constexpr int H      = 1024;
constexpr int CTX    = 300;
constexpr int CTXP   = 320;
constexpr int ATT    = 100;
constexpr int ATTP   = 128;
constexpr int DANS   = 556;   // LEX_D + 2*NT_D (ans-embed part of decoder input)
constexpr int DANSP  = 576;   // padded to x32

// LDS row strides (u16 units) chosen for 16B-aligned b128 fragments + lane bank spread
constexpr int SH_ENC = 1040;  // 1024 + TDM pad (4 DW every 256 DW)
constexpr int SH_DEC = 1032;  // software padded
constexpr int SA_DEC = 584;
constexpr int SC_DEC = 328;

// ---------------- bf16 helpers ----------------
__device__ inline u16 f2bf(float f) {
    unsigned u = __float_as_uint(f);
    unsigned r = ((u >> 16) & 1u) + 0x7fffu;   // round-to-nearest-even
    return (u16)((u + r) >> 16);
}
__device__ inline float b2f(u16 h) { return __uint_as_float(((unsigned)h) << 16); }
__device__ inline float sigm(float x) { return 1.f / (1.f + __expf(-x)); }

// ---------------- WMMA fragment helpers (contiguous k = 16*half + s order, wave32) ----------------
union Frag { v16bf v; uint4 q[2]; };

// A-matrix 16x32 bf16: lane&15 = row M; lane>>4 selects K 0..15 / 16..31
__device__ inline Frag ldA(const u16* row, int k0, int half) {
    Frag f;
    f.q[0] = *(const uint4*)(row + k0 + half * 16);
    f.q[1] = *(const uint4*)(row + k0 + half * 16 + 8);
    return f;
}
// B-matrix 32x16 from row-major W[N][K]: lane&15 = col N; lane>>4 selects K 0..15 / 16..31
__device__ inline Frag ldB(const u16* p) {
    Frag f;
    f.q[0] = *(const uint4*)(p);
    f.q[1] = *(const uint4*)(p + 8);
    return f;
}
__device__ inline v8f wmma_bf16(const Frag& a, const Frag& b, v8f c) {
    return __builtin_amdgcn_wmma_f32_16x16x32_bf16(false, a.v, false, b.v,
                                                   (short)0, c, false, false);
}

// ---------------- TDM: 2D tile (rows x dim0 bf16) global -> LDS ----------------
// D# per CDNA5 ISA. midpad: pad_enable, interval=256DW, amount=4DW
// => LDS row stride 1040 u16, +8 u16 after k>=512 inside a row (16B aligned).
__device__ inline void tdm_load_2d(unsigned lds, unsigned long long ga,
                                   unsigned dim0, unsigned rows, unsigned stride0,
                                   bool midpad) {
    v4u g0;
    g0[0] = 1u;                                       // count=1, user mode
    g0[1] = lds;                                      // lds_addr
    g0[2] = (unsigned)ga;                             // global_addr[31:0]
    g0[3] = (unsigned)((ga >> 32) & 0x1FFFFFFu) | (2u << 30);  // addr[56:32], type=2
    v8u g1;
    unsigned w0 = 1u << 16;                           // data_size = 2 bytes
    if (midpad) w0 |= (1u << 20) | (7u << 22) | (3u << 25);  // pad_en, 256DW interval, 4DW pad
    g1[0] = w0;
    g1[1] = (dim0 & 0xFFFFu) << 16;                   // tensor_dim0 lo
    g1[2] = (dim0 >> 16) | ((rows & 0xFFFFu) << 16);  // tensor_dim0 hi | tensor_dim1 lo
    g1[3] = ((dim0 & 0xFFFFu) << 16);                 // tensor_dim1 hi(0) | tile_dim0
    g1[4] = rows & 0xFFFFu;                           // tile_dim1 | tile_dim2(0)
    g1[5] = stride0;                                  // tensor_dim0_stride lo (elements)
    g1[6] = 0u;                                       // stride hi | dim1_stride lo
    g1[7] = 0u;
    asm volatile("tensor_load_to_lds %0, %1" : : "s"(g0), "s"(g1) : "memory");
}

// ---------------- async global -> LDS tile copy (per-lane b128 transfers) ----------------
__device__ inline void async_tile(u16* ldsBase, int ldsStride, const u16* g, long gStride,
                                  int cols, int tid) {
    int per = cols >> 3;                   // 16B transfers per row
    int tot = 16 * per;
    for (int i = tid; i < tot; i += 128) {
        int row = i / per, c8 = (i - row * per) << 3;
        unsigned lds = (unsigned)(uintptr_t)(ldsBase + row * ldsStride + c8);
        unsigned long long ga = (unsigned long long)(uintptr_t)(g + (long)row * gStride + c8);
        asm volatile("global_load_async_to_lds_b128 %0, %1, off"
                     : : "v"(lds), "v"(ga) : "memory");
    }
}
__device__ inline void wait_async0() {
    asm volatile("s_wait_asynccnt 0x0" ::: "memory");
}

// ---------------- pipelined WMMA K-segment: acc += A(16xK from LDS) @ W(64xK)^T ----------------
// aRow: per-lane LDS row base. wc*: W + row*ldw + half*16 (per-lane). MIDPAD: +8 u16 for k>=512.
template<int K, bool MIDPAD>
__device__ inline void mma_seg(v8f acc[4], const u16* aRow,
                               const u16* w0c, const u16* w1c,
                               const u16* w2c, const u16* w3c, int half) {
    const u16* wc[4] = {w0c, w1c, w2c, w3c};
    Frag a0 = ldA(aRow, 0, half);
    Frag b0[4];
#pragma unroll
    for (int s = 0; s < 4; ++s) b0[s] = ldB(wc[s]);
#pragma unroll
    for (int k0 = 0; k0 < K; k0 += 32) {
        const int k1 = k0 + 32;
        Frag a1 = a0;
        Frag b1[4] = {b0[0], b0[1], b0[2], b0[3]};
        if (k1 < K) {                                  // prefetch next chunk's fragments
            a1 = ldA(aRow + ((MIDPAD && k1 >= 512) ? 8 : 0), k1, half);
#pragma unroll
            for (int s = 0; s < 4; ++s) b1[s] = ldB(wc[s] + k1);
        }
#pragma unroll
        for (int s = 0; s < 4; ++s) acc[s] = wmma_bf16(a0, b0[s], acc[s]);
        a0 = a1;
#pragma unroll
        for (int s = 0; s < 4; ++s) b0[s] = b1[s];
    }
}

// ---------------- fp32 -> padded bf16 conversion ----------------
__global__ void k_cvt(const float* __restrict__ src, int sld, int c0, int R, int C,
                      u16* __restrict__ dst, int dld, int Rp) {
    long i = (long)blockIdx.x * 256 + threadIdx.x;
    long n = (long)Rp * dld;
    if (i >= n) return;
    int r = (int)(i / dld), c = (int)(i % dld);
    float v = (r < R && c < C) ? src[(size_t)r * sld + c0 + c] : 0.f;
    dst[i] = f2bf(v);
}

// ---------------- embedding gathers ----------------
__global__ void k_gather_src(const int* __restrict__ seq, const float* __restrict__ emb,
                             u16* __restrict__ X) {
    long i = (long)blockIdx.x * 256 + threadIdx.x;
    long n = (long)B * S_SRC * LDP;
    if (i >= n) return;
    int d = (int)(i % LDP);
    long r = i / LDP;
    int tok = seq[r];
    float v = (d < LEX_D) ? emb[(size_t)tok * LEX_D + d] : 0.f;
    X[i] = f2bf(v);
}

__global__ void k_gather_ans(const int* __restrict__ trg, const int* __restrict__ pn,
                             const int* __restrict__ pl, const float* __restrict__ nt,
                             const float* __restrict__ lex, u16* __restrict__ X) {
    long i = (long)blockIdx.x * 256 + threadIdx.x;
    long n = (long)B * S_TRG * DANSP;
    if (i >= n) return;
    int d = (int)(i % DANSP);
    long r = i / DANSP;
    float v = 0.f;
    if (d < NT_D)            v = nt[(size_t)trg[r] * NT_D + d];
    else if (d < 2 * NT_D)   v = nt[(size_t)pn[r] * NT_D + (d - NT_D)];
    else if (d < DANS)       v = lex[(size_t)pl[r] * LEX_D + (d - 2 * NT_D)];
    X[i] = f2bf(v);
}

// ---------------- generic pipelined WMMA GEMM: OUT[MxN] = A[MxK] @ W[NxK]^T + bias ----------------
__global__ void __launch_bounds__(128)
k_gemm(const u16* __restrict__ A, int lda, const u16* __restrict__ W, int ldw,
       const float* __restrict__ bias, float* __restrict__ outF, u16* __restrict__ outB,
       int ldo, int M, int Nreal, int K, int act) {
    int tid = threadIdx.x, lane = tid & 31, wave = tid >> 5;
    int m0 = blockIdx.y * 64 + wave * 16;
    int n0 = blockIdx.x * 64;
    int half = lane >> 4, mr = lane & 15, nc = lane & 15;

    const u16* arow = A + (size_t)(m0 + mr) * lda;
    const u16* wc[4];
#pragma unroll
    for (int s = 0; s < 4; ++s) wc[s] = W + (size_t)(n0 + s * 16 + nc) * ldw + half * 16;

    v8f acc[4] = {};
    Frag a0 = ldA(arow, 0, half);
    Frag b0[4];
#pragma unroll
    for (int s = 0; s < 4; ++s) b0[s] = ldB(wc[s]);
    for (int k0 = 0; k0 < K; k0 += 32) {
        int k1 = k0 + 32;
        Frag a1 = a0;
        Frag b1[4] = {b0[0], b0[1], b0[2], b0[3]};
        if (k1 < K) {
            a1 = ldA(arow, k1, half);
            __builtin_prefetch(arow + k1 + 32, 0, 3);      // global_prefetch
#pragma unroll
            for (int s = 0; s < 4; ++s) b1[s] = ldB(wc[s] + k1);
        }
#pragma unroll
        for (int s = 0; s < 4; ++s) acc[s] = wmma_bf16(a0, b0[s], acc[s]);
        a0 = a1;
#pragma unroll
        for (int s = 0; s < 4; ++s) b0[s] = b1[s];
    }
#pragma unroll
    for (int s = 0; s < 4; ++s) {
        int n = n0 + s * 16 + nc;
        if (n < Nreal) {
            float bv = bias ? bias[n] : 0.f;
#pragma unroll
            for (int r = 0; r < 8; ++r) {
                int row = m0 + half * 8 + r;
                if (row < M) {
                    float v = acc[s][r] + bv;
                    if (act == 1) v = v > 0.f ? v : 0.f;
                    else if (act == 2) v = tanhf(v);
                    if (outF) outF[(size_t)row * ldo + n] = v;
                    if (outB) outB[(size_t)row * ldo + n] = f2bf(v);
                }
            }
        }
    }
}

// ---------------- fused encoder LSTM step (TDM-staged activations) ----------------
__global__ void __launch_bounds__(128)
k_enc_step(const u16* __restrict__ hin, u16* __restrict__ hout, float* __restrict__ c,
           const u16* __restrict__ Whh, const u16* __restrict__ Wih,
           const u16* __restrict__ Xsrc, const float* __restrict__ bias,
           const int* __restrict__ lens, u16* __restrict__ srcH, int t) {
    __shared__ u16 sH[16 * SH_ENC];
    __shared__ u16 sX[16 * LDP];
    __shared__ float zb[4][16][64];
    int tid = threadIdx.x, lane = tid & 31, wave = tid >> 5;
    int m0 = blockIdx.y * 16, n0 = blockIdx.x * 64;
    int half = lane >> 4, mr = lane & 15, nc = lane & 15;
    int gofs = wave * H;

    if (wave == 0) {            // TDM issues once (EXEC-independent DMA), wave-uniform D#
        tdm_load_2d((unsigned)(uintptr_t)&sH[0],
                    (unsigned long long)(uintptr_t)(hin + (size_t)m0 * H),
                    H, 16, H, /*midpad*/true);
        tdm_load_2d((unsigned)(uintptr_t)&sX[0],
                    (unsigned long long)(uintptr_t)(Xsrc + ((size_t)m0 * S_SRC + t) * LDP),
                    LDP, 16, S_SRC * LDP, /*midpad*/false);
        __builtin_amdgcn_s_wait_tensorcnt(0);
    }
    __syncthreads();

    v8f acc[4] = {};
    mma_seg<H, true>(acc, sH + mr * SH_ENC,
        Whh + (size_t)(gofs + n0 +  0 + nc) * H + half * 16,
        Whh + (size_t)(gofs + n0 + 16 + nc) * H + half * 16,
        Whh + (size_t)(gofs + n0 + 32 + nc) * H + half * 16,
        Whh + (size_t)(gofs + n0 + 48 + nc) * H + half * 16, half);
    mma_seg<LDP, false>(acc, sX + mr * LDP,
        Wih + (size_t)(gofs + n0 +  0 + nc) * LDP + half * 16,
        Wih + (size_t)(gofs + n0 + 16 + nc) * LDP + half * 16,
        Wih + (size_t)(gofs + n0 + 32 + nc) * LDP + half * 16,
        Wih + (size_t)(gofs + n0 + 48 + nc) * LDP + half * 16, half);

#pragma unroll
    for (int s = 0; s < 4; ++s) {
        float bv = bias[gofs + n0 + s * 16 + nc];
#pragma unroll
        for (int r = 0; r < 8; ++r)
            zb[wave][half * 8 + r][s * 16 + nc] = acc[s][r] + bv;
    }
    __syncthreads();
    for (int e = tid; e < 16 * 64; e += 128) {
        int r = e >> 6, col = e & 63;
        int b = m0 + r, j = n0 + col;
        float zi = zb[0][r][col], zf = zb[1][r][col], zg = zb[2][r][col], zo = zb[3][r][col];
        float cold = c[(size_t)b * H + j];
        float c2 = sigm(zf) * cold + sigm(zi) * tanhf(zg);
        float h2 = sigm(zo) * tanhf(c2);
        bool valid = t < lens[b];
        if (valid) c[(size_t)b * H + j] = c2;
        u16 hold = sH[r * SH_ENC + j + (j >= 512 ? 8 : 0)];   // previous h from staged tile
        hout[(size_t)b * H + j] = valid ? f2bf(h2) : hold;
        srcH[((size_t)b * S_SRC + t) * H + j] = valid ? f2bf(h2) : (u16)0;
    }
}

// ---------------- per-step attention (VALU; ~1% of flops) ----------------
__global__ void __launch_bounds__(128)
k_attn(const u16* __restrict__ h, const float* __restrict__ Wa, const float* __restrict__ ba,
       const u16* __restrict__ qkey, const u16* __restrict__ qval,
       const int* __restrict__ lens, u16* __restrict__ ctxb) {
    __shared__ float hf[H];
    __shared__ float ak[ATT];
    __shared__ float wts[S_SRC];
    __shared__ float red[128];
    int b = blockIdx.x, tid = threadIdx.x;
    for (int j = tid; j < H; j += 128) hf[j] = b2f(h[(size_t)b * H + j]);
    __syncthreads();
    for (int k = tid; k < ATT; k += 128) {
        float s = ba[k];
        const float* wr = Wa + (size_t)k * H;
        for (int j = 0; j < H; ++j) s += hf[j] * wr[j];
        ak[k] = tanhf(s);
    }
    __syncthreads();
    const u16* qr = qkey + ((size_t)b * S_SRC + tid) * ATTP;
    float e = 0.f;
    for (int k = 0; k < ATT; ++k) e += b2f(qr[k]) * ak[k];
    e = (tid < lens[b]) ? e : -1e9f;
    red[tid] = e; __syncthreads();
    for (int o = 64; o > 0; o >>= 1) { if (tid < o) red[tid] = fmaxf(red[tid], red[tid + o]); __syncthreads(); }
    float mx = red[0]; __syncthreads();
    float ex = __expf(e - mx);
    red[tid] = ex; __syncthreads();
    for (int o = 64; o > 0; o >>= 1) { if (tid < o) red[tid] += red[tid + o]; __syncthreads(); }
    float inv = 1.f / red[0];
    wts[tid] = ex * inv; __syncthreads();
    for (int d = tid; d < CTXP; d += 128) {
        float a = 0.f;
        if (d < CTX)
            for (int s = 0; s < S_SRC; ++s)
                a += wts[s] * b2f(qval[((size_t)b * S_SRC + s) * CTXP + d]);
        ctxb[(size_t)b * CTXP + d] = f2bf(a);
    }
}

// ---------------- fused decoder LSTM step (async-to-LDS staged activations) ----------------
__global__ void __launch_bounds__(128)
k_dec_step(const u16* __restrict__ hin, u16* __restrict__ hout, float* __restrict__ c,
           const u16* __restrict__ Whh, const u16* __restrict__ Wans, const u16* __restrict__ Wctx,
           const u16* __restrict__ ansX, const u16* __restrict__ ctxb,
           const float* __restrict__ bias, float* __restrict__ out, int t) {
    __shared__ u16 sH[16 * SH_DEC];
    __shared__ u16 sA[16 * SA_DEC];
    __shared__ u16 sC[16 * SC_DEC];
    __shared__ float zb[4][16][64];
    int tid = threadIdx.x, lane = tid & 31, wave = tid >> 5;
    int m0 = blockIdx.y * 16, n0 = blockIdx.x * 64;
    int half = lane >> 4, mr = lane & 15, nc = lane & 15;
    int gofs = wave * H;

    async_tile(sH, SH_DEC, hin + (size_t)m0 * H, H, H, tid);
    async_tile(sA, SA_DEC, ansX + ((size_t)m0 * S_TRG + t) * DANSP, (long)S_TRG * DANSP, DANSP, tid);
    async_tile(sC, SC_DEC, ctxb + (size_t)m0 * CTXP, CTXP, CTXP, tid);
    wait_async0();
    __syncthreads();

    v8f acc[4] = {};
    mma_seg<H, false>(acc, sH + mr * SH_DEC,
        Whh + (size_t)(gofs + n0 +  0 + nc) * H + half * 16,
        Whh + (size_t)(gofs + n0 + 16 + nc) * H + half * 16,
        Whh + (size_t)(gofs + n0 + 32 + nc) * H + half * 16,
        Whh + (size_t)(gofs + n0 + 48 + nc) * H + half * 16, half);
    mma_seg<DANSP, false>(acc, sA + mr * SA_DEC,
        Wans + (size_t)(gofs + n0 +  0 + nc) * DANSP + half * 16,
        Wans + (size_t)(gofs + n0 + 16 + nc) * DANSP + half * 16,
        Wans + (size_t)(gofs + n0 + 32 + nc) * DANSP + half * 16,
        Wans + (size_t)(gofs + n0 + 48 + nc) * DANSP + half * 16, half);
    mma_seg<CTXP, false>(acc, sC + mr * SC_DEC,
        Wctx + (size_t)(gofs + n0 +  0 + nc) * CTXP + half * 16,
        Wctx + (size_t)(gofs + n0 + 16 + nc) * CTXP + half * 16,
        Wctx + (size_t)(gofs + n0 + 32 + nc) * CTXP + half * 16,
        Wctx + (size_t)(gofs + n0 + 48 + nc) * CTXP + half * 16, half);

#pragma unroll
    for (int s = 0; s < 4; ++s) {
        float bv = bias[gofs + n0 + s * 16 + nc];
#pragma unroll
        for (int r = 0; r < 8; ++r)
            zb[wave][half * 8 + r][s * 16 + nc] = acc[s][r] + bv;
    }
    __syncthreads();
    for (int e = tid; e < 16 * 64; e += 128) {
        int r = e >> 6, col = e & 63;
        int b = m0 + r, j = n0 + col;
        float zi = zb[0][r][col], zf = zb[1][r][col], zg = zb[2][r][col], zo = zb[3][r][col];
        float cold = c[(size_t)b * H + j];
        float c2 = sigm(zf) * cold + sigm(zi) * tanhf(zg);
        float h2 = sigm(zo) * tanhf(c2);
        c[(size_t)b * H + j] = c2;
        hout[(size_t)b * H + j] = f2bf(h2);
        out[((size_t)b * S_TRG + t) * H + j] = h2;
    }
}

// ---------------- host launch ----------------
extern "C" void kernel_launch(void* const* d_in, const int* in_sizes, int n_in,
                              void* d_out, int out_size, void* d_ws, size_t ws_size,
                              hipStream_t stream) {
    (void)in_sizes; (void)n_in; (void)out_size; (void)ws_size;
    const int*   src_seqs = (const int*)d_in[0];
    const int*   src_len  = (const int*)d_in[1];
    const int*   trg_nt   = (const int*)d_in[2];
    const int*   par_nt   = (const int*)d_in[3];
    const int*   par_lex  = (const int*)d_in[4];
    const float* lex_emb  = (const float*)d_in[5];
    const float* nt_emb   = (const float*)d_in[6];
    const float* enc_Wih  = (const float*)d_in[7];
    const float* enc_Whh  = (const float*)d_in[8];
    const float* enc_b    = (const float*)d_in[9];
    const float* dec_Wih  = (const float*)d_in[10];
    const float* dec_Whh  = (const float*)d_in[11];
    const float* dec_b    = (const float*)d_in[12];
    const float* Wh1 = (const float*)d_in[13]; const float* bh1 = (const float*)d_in[14];
    const float* Wh2 = (const float*)d_in[15]; const float* bh2 = (const float*)d_in[16];
    const float* Wc1 = (const float*)d_in[17]; const float* bc1 = (const float*)d_in[18];
    const float* Wc2 = (const float*)d_in[19]; const float* bc2 = (const float*)d_in[20];
    const float* Wa  = (const float*)d_in[21]; const float* ba  = (const float*)d_in[22];
    const float* Wqk = (const float*)d_in[23]; const float* bqk = (const float*)d_in[24];
    const float* Wqv = (const float*)d_in[25]; const float* bqv = (const float*)d_in[26];
    float* out = (float*)d_out;

    char* p = (char*)d_ws;
    auto alloc = [&](size_t bytes) -> void* {
        void* r = (void*)p; p += (bytes + 255) & ~(size_t)255; return r;
    };
    u16* Xsrc  = (u16*)alloc((size_t)B * S_SRC * LDP * 2);
    u16* ansX  = (u16*)alloc((size_t)B * S_TRG * DANSP * 2);
    u16* srcH  = (u16*)alloc((size_t)B * S_SRC * H * 2);
    u16* qkey  = (u16*)alloc((size_t)B * S_SRC * ATTP * 2);
    u16* qval  = (u16*)alloc((size_t)B * S_SRC * CTXP * 2);
    u16* Wihe  = (u16*)alloc((size_t)4 * H * LDP * 2);
    u16* Whhe  = (u16*)alloc((size_t)4 * H * H * 2);
    u16* Wansb = (u16*)alloc((size_t)4 * H * DANSP * 2);
    u16* Wctxb = (u16*)alloc((size_t)4 * H * CTXP * 2);
    u16* Whhd  = (u16*)alloc((size_t)4 * H * H * 2);
    u16* Wh1b  = (u16*)alloc((size_t)2 * H * H * 2);
    u16* Wh2b  = (u16*)alloc((size_t)H * 2 * H * 2);
    u16* Wc1b  = (u16*)alloc((size_t)2 * H * H * 2);
    u16* Wc2b  = (u16*)alloc((size_t)H * 2 * H * 2);
    u16* Wqkb  = (u16*)alloc((size_t)ATTP * H * 2);
    u16* Wqvb  = (u16*)alloc((size_t)CTXP * H * 2);
    u16* h0    = (u16*)alloc((size_t)B * H * 2);
    u16* h1    = (u16*)alloc((size_t)B * H * 2);
    u16* cTb   = (u16*)alloc((size_t)B * H * 2);
    u16* hd0   = (u16*)alloc((size_t)B * H * 2);
    u16* hd1   = (u16*)alloc((size_t)B * H * 2);
    u16* ctxb  = (u16*)alloc((size_t)B * CTXP * 2);
    u16* t1b   = (u16*)alloc((size_t)B * 2 * H * 2);
    float* c_enc = (float*)alloc((size_t)B * H * 4);
    float* c_dec = (float*)alloc((size_t)B * H * 4);

    hipMemsetAsync(h0, 0, (size_t)B * H * 2, stream);
    hipMemsetAsync(c_enc, 0, (size_t)B * H * 4, stream);

    auto cvt = [&](const float* s, int sld, int c0, int R, int C, u16* d, int dld, int Rp) {
        long n = (long)Rp * dld;
        k_cvt<<<(unsigned)((n + 255) / 256), 256, 0, stream>>>(s, sld, c0, R, C, d, dld, Rp);
    };
    cvt(enc_Wih, LEX_D, 0,    4 * H, LEX_D, Wihe,  LDP,   4 * H);
    cvt(enc_Whh, H,     0,    4 * H, H,     Whhe,  H,     4 * H);
    cvt(dec_Wih, 856,   0,    4 * H, DANS,  Wansb, DANSP, 4 * H);
    cvt(dec_Wih, 856,   DANS, 4 * H, CTX,   Wctxb, CTXP,  4 * H);
    cvt(dec_Whh, H,     0,    4 * H, H,     Whhd,  H,     4 * H);
    cvt(Wh1, H,     0, 2 * H, H,     Wh1b, H,     2 * H);
    cvt(Wh2, 2 * H, 0, H,     2 * H, Wh2b, 2 * H, H);
    cvt(Wc1, H,     0, 2 * H, H,     Wc1b, H,     2 * H);
    cvt(Wc2, 2 * H, 0, H,     2 * H, Wc2b, 2 * H, H);
    cvt(Wqk, H, 0, ATT, H, Wqkb, H, ATTP);
    cvt(Wqv, H, 0, CTX, H, Wqvb, H, CTXP);

    {
        long n = (long)B * S_SRC * LDP;
        k_gather_src<<<(unsigned)((n + 255) / 256), 256, 0, stream>>>(src_seqs, lex_emb, Xsrc);
        n = (long)B * S_TRG * DANSP;
        k_gather_ans<<<(unsigned)((n + 255) / 256), 256, 0, stream>>>(trg_nt, par_nt, par_lex,
                                                                      nt_emb, lex_emb, ansX);
    }

    for (int t = 0; t < S_SRC; ++t) {
        const u16* hi = (t & 1) ? h1 : h0;
        u16*       ho = (t & 1) ? h0 : h1;
        k_enc_step<<<dim3(H / 64, B / 16), 128, 0, stream>>>(hi, ho, c_enc, Whhe, Wihe,
                                                             Xsrc, enc_b, src_len, srcH, t);
    }
    u16* hTb = h0;   // S_SRC even -> final h in h0

    k_gemm<<<dim3(ATTP / 64, (B * S_SRC) / 64), 128, 0, stream>>>(
        srcH, H, Wqkb, H, bqk, nullptr, qkey, ATTP, B * S_SRC, ATT, H, /*tanh*/2);
    k_gemm<<<dim3(CTXP / 64, (B * S_SRC) / 64), 128, 0, stream>>>(
        srcH, H, Wqvb, H, bqv, nullptr, qval, CTXP, B * S_SRC, CTX, H, 0);

    cvt(c_enc, H, 0, B, H, cTb, H, B);
    k_gemm<<<dim3(2 * H / 64, B / 64), 128, 0, stream>>>(
        hTb, H, Wh1b, H, bh1, nullptr, t1b, 2 * H, B, 2 * H, H, /*relu*/1);
    k_gemm<<<dim3(H / 64, B / 64), 128, 0, stream>>>(
        t1b, 2 * H, Wh2b, 2 * H, bh2, nullptr, hd0, H, B, H, 2 * H, 0);
    k_gemm<<<dim3(2 * H / 64, B / 64), 128, 0, stream>>>(
        cTb, H, Wc1b, H, bc1, nullptr, t1b, 2 * H, B, 2 * H, H, /*relu*/1);
    k_gemm<<<dim3(H / 64, B / 64), 128, 0, stream>>>(
        t1b, 2 * H, Wc2b, 2 * H, bc2, c_dec, nullptr, H, B, H, 2 * H, 0);

    for (int t = 0; t < S_TRG; ++t) {
        const u16* hi = (t & 1) ? hd1 : hd0;
        u16*       ho = (t & 1) ? hd0 : hd1;
        k_attn<<<B, 128, 0, stream>>>(hi, Wa, ba, qkey, qval, src_len, ctxb);
        k_dec_step<<<dim3(H / 64, B / 16), 128, 0, stream>>>(hi, ho, c_dec, Whhd, Wansb,
                                                             Wctxb, ansX, ctxb, dec_b, out, t);
    }
}
